// WinBasedAttention_89515708383550
// MI455X (gfx1250) — compile-verified, matrix operand order checked by reading, not run on hardware
//
#include <hip/hip_runtime.h>
#include <hip/hip_bf16.h>

typedef _Float16 v16h __attribute__((ext_vector_type(16)));
typedef float    v8f  __attribute__((ext_vector_type(8)));
typedef unsigned int u32x4 __attribute__((ext_vector_type(4)));
typedef int          i32x8 __attribute__((ext_vector_type(8)));
typedef int          i32x4 __attribute__((ext_vector_type(4)));

#define NH_  8
#define HD_  24
#define C_   192
#define NT_  64
#define HRES 256
#define WRES 256

// workspace layout (bytes)
#define OFF_BIAS   0                        // 8*64*64 f32 = 131072
#define OFF_WQKV   131072                   // 576*192 f16 = 221184
#define OFF_WPROJ  (131072 + 221184)        // 192*192 f16 = 73728
#define OFF_QKVB   (131072 + 221184 + 73728)// 576 f32

// ---------------------------------------------------------------------------
// WMMA helpers (gfx1250, wave32)
// ---------------------------------------------------------------------------
__device__ __forceinline__ v8f wmma_f16(v16h a, v16h b, v8f c) {
  return __builtin_amdgcn_wmma_f32_16x16x32_f16(false, a, false, b, (short)0, c,
                                                false, false);
}

// Fragment from a row-major f16 matrix: lane supplies one row, K pattern is
// [k0+kb+0..7] then [k0+kb+16..23] with kb = (lane>=16) ? 8 : 0.
__device__ __forceinline__ v16h frag_rows(const _Float16* __restrict__ m,
                                          int ld, int row, int k0, int kb) {
  const _Float16* p = m + row * ld + k0 + kb;
  v16h f;
#pragma unroll
  for (int e = 0; e < 8; ++e) f[e] = p[e];
#pragma unroll
  for (int e = 0; e < 8; ++e) f[8 + e] = p[16 + e];
  return f;
}

__device__ __forceinline__ v16h frag_rows_guard(const _Float16* __restrict__ m,
                                                int ld, int row, int k0, int kb,
                                                bool valid) {
  v16h f;
  if (valid) {
    f = frag_rows(m, ld, row, k0, kb);
  } else {
#pragma unroll
    for (int e = 0; e < 16; ++e) f[e] = (_Float16)0.0f;
  }
  return f;
}

// B fragment when the source is K-major (V: tokens x dims): lane holds one
// column (dim), reads strided over K (tokens).
__device__ __forceinline__ v16h frag_cols(const _Float16* __restrict__ m,
                                          int ld, int k0, int col, int kb) {
  const _Float16* p = m + (k0 + kb) * ld + col;
  v16h f;
#pragma unroll
  for (int e = 0; e < 8; ++e) f[e] = p[e * ld];
#pragma unroll
  for (int e = 0; e < 8; ++e) f[8 + e] = p[(16 + e) * ld];
  return f;
}

// ---------------------------------------------------------------------------
// Prologue 1: relative-position-bias table via the tiny CPB MLP (shared by
// all 4096 windows -> computed once into ws).
// ---------------------------------------------------------------------------
__global__ __launch_bounds__(256) void cpb_bias_kernel(
    const float* __restrict__ w1, const float* __restrict__ b1,
    const float* __restrict__ w2, float* __restrict__ biasTab) {
  __shared__ float table[225 * NH_];
  const int tid = threadIdx.x;
  if (tid < 225) {
    int i = tid / 15, j = tid % 15;
    float c0 = (float)(i - 7) * (8.0f / 7.0f);
    float c1 = (float)(j - 7) * (8.0f / 7.0f);
    float t0 = copysignf(__log2f(fabsf(c0) + 1.0f) * (1.0f / 3.0f), c0);
    float t1 = copysignf(__log2f(fabsf(c1) + 1.0f) * (1.0f / 3.0f), c1);
    float o[NH_];
#pragma unroll
    for (int hh = 0; hh < NH_; ++hh) o[hh] = 0.0f;
    for (int jj = 0; jj < 512; ++jj) {
      float hv = fmaf(t0, w1[jj], fmaf(t1, w1[512 + jj], b1[jj]));
      hv = fmaxf(hv, 0.0f);
#pragma unroll
      for (int hh = 0; hh < NH_; ++hh) o[hh] += hv * w2[jj * NH_ + hh];
    }
#pragma unroll
    for (int hh = 0; hh < NH_; ++hh) table[tid * NH_ + hh] = o[hh];
  }
  __syncthreads();
  for (int idx = tid; idx < NH_ * NT_ * NT_; idx += 256) {
    int hh = idx >> 12;
    int m = (idx >> 6) & 63;
    int n = idx & 63;
    int rel0 = (m >> 3) - (n >> 3) + 7;
    int rel1 = (m & 7) - (n & 7) + 7;
    float v = table[(rel0 * 15 + rel1) * NH_ + hh];
    biasTab[idx] = 16.0f / (1.0f + __expf(-v));
  }
}

// ---------------------------------------------------------------------------
// Prologue 2: weights -> f16, fused qkv bias [q_bias, 0, v_bias].
// ---------------------------------------------------------------------------
__global__ void prep_weights_kernel(const float* __restrict__ qkv_w,
                                    const float* __restrict__ proj_w,
                                    const float* __restrict__ q_bias,
                                    const float* __restrict__ v_bias,
                                    _Float16* __restrict__ WqkvH,
                                    _Float16* __restrict__ WprojH,
                                    float* __restrict__ qkvb) {
  int idx = blockIdx.x * blockDim.x + threadIdx.x;
  const int n1 = 576 * 192, n2 = 192 * 192;
  if (idx < n1) {
    WqkvH[idx] = (_Float16)qkv_w[idx];
  } else if (idx < n1 + n2) {
    WprojH[idx - n1] = (_Float16)proj_w[idx - n1];
  } else if (idx < n1 + n2 + 576) {
    int r = idx - n1 - n2;
    qkvb[r] = (r < 192) ? q_bias[r] : (r < 384) ? 0.0f : v_bias[r - 384];
  }
}

// ---------------------------------------------------------------------------
// Fused window attention: one window per workgroup, one wave32 per head.
// ---------------------------------------------------------------------------
__global__ __launch_bounds__(256) void winattn_main(
    const float* __restrict__ x, const _Float16* __restrict__ Wqkv,
    const _Float16* __restrict__ Wproj, const float* __restrict__ qkvb,
    const float* __restrict__ biasTab, const float* __restrict__ logit_scale,
    const float* __restrict__ proj_b, float* __restrict__ out) {
  __shared__ float    sXf32[NT_ * C_];     // TDM dest, channel-major [c][t]; residual
  __shared__ _Float16 sX[NT_ * C_];        // token-major f16; reused for attn out
  __shared__ _Float16 sQ[NH_][NT_ * 32];   // normalized+scaled q, K-padded to 32
  __shared__ _Float16 sK[NH_][NT_ * 32];   // normalized k
  __shared__ _Float16 sV[NH_][NT_ * 32];   // v (dims 24..31 zero)
  __shared__ _Float16 sP[NH_][NT_ * NT_];  // softmax probabilities

  const int tid  = threadIdx.x;
  const int wave = tid >> 5;
  const int lane = tid & 31;
  const int l15  = lane & 15;
  const int kb   = (lane & 16) ? 8 : 0;  // K-offset half of the fragment
  const int hi8  = (lane & 16) ? 8 : 0;  // row offset for C/D fragments
  const int h    = wave;                  // head == wave

  const int wid  = blockIdx.x;
  const int bb   = wid >> 10;
  const int wly  = wid & 1023;
  const int row0 = (wly >> 5) * 8;
  const int col0 = (wly & 31) * 8;

  // ---- Phase A: Tensor Data Mover pulls the whole (8 x 8 x 192ch) window
  //      tile into LDS, channel-major (x fastest, then y, then c), in ONE
  //      instruction. D# per CDNA5 ISA 8.3-8.6; wave 0 issues, waits
  //      TENSORcnt, then the workgroup barriers.
  if (wave == 0) {
    unsigned long long gaddr =
        (unsigned long long)(uintptr_t)(x + (((size_t)bb * C_) * HRES + row0) *
                                                WRES + col0);
    unsigned int ldsa = (unsigned int)(uintptr_t)(&sXf32[0]);

    u32x4 g0;
    g0[0] = 1u;                                   // count=1, not restore/gather
    g0[1] = ldsa;                                 // lds_addr
    g0[2] = (unsigned int)(gaddr & 0xFFFFFFFFu);  // global_addr[31:0]
    g0[3] = (unsigned int)((gaddr >> 32) & 0x01FFFFFFu) | 0x80000000u; // type=2

    i32x8 g1;
    g1[0] = (int)(2u << 16);          // workgroup_mask=0, data_size=4B
    g1[1] = (int)(256u << 16);        // tensor_dim0 lo16 (=256) at bits[63:48]
    g1[2] = (int)(256u << 16);        // dim0 hi16=0 | tensor_dim1 lo16 (=256)
    g1[3] = (int)(8u << 16);          // dim1 hi16=0 | tile_dim0 = 8
    g1[4] = (int)(8u | (192u << 16)); // tile_dim1 = 8, tile_dim2 = 192
    g1[5] = 256;                      // tensor_dim0_stride[31:0] = 256
    g1[6] = 0;                        // stride0 hi | stride1 lo16 (65536->0)
    g1[7] = 1;                        // tensor_dim1_stride[47:16] = 1 (65536)

    i32x4 g2;
    g2[0] = 768;      // tensor_dim2 (B*C planes)
    g2[1] = 0;        // tensor_dim3 (unused)
    g2[2] = 65536;    // tensor_dim2_stride[31:0] = H*W
    g2[3] = 0;        // stride2 hi16 = 0, tile_dim3 = 0

    i32x4 g3;
    g3[0] = 0; g3[1] = 0; g3[2] = 0; g3[3] = 0;

    i32x8 gx;         // extra operand of the 6-arg builtin: zero-filled
    gx[0] = 0; gx[1] = 0; gx[2] = 0; gx[3] = 0;
    gx[4] = 0; gx[5] = 0; gx[6] = 0; gx[7] = 0;

    __builtin_amdgcn_tensor_load_to_lds(g0, g1, g2, g3, gx, 0);
    __builtin_amdgcn_s_wait_tensorcnt(0);
  }
  __syncthreads();

  // ---- Phase A2: f32 (channel-major) -> f16 token-major copy for WMMA A ---
  for (int q4 = tid; q4 < (NT_ * C_) / 4; q4 += 256) {
    float4 v4 = *reinterpret_cast<const float4*>(&sXf32[q4 * 4]);
    int l = q4 * 4;
    int c = l >> 6;       // channel
    int t = l & 63;       // token (4 consecutive)
    sX[(t + 0) * C_ + c] = (_Float16)v4.x;
    sX[(t + 1) * C_ + c] = (_Float16)v4.y;
    sX[(t + 2) * C_ + c] = (_Float16)v4.z;
    sX[(t + 3) * C_ + c] = (_Float16)v4.w;
  }
  __syncthreads();

  const float scale_h = __expf(fminf(logit_scale[h], 4.6051701860f)); // ln(100)

  // ---- Phase B: QKV projection for this wave's head, cosine-normalize ----
  {
    const int wrow = h * HD_;  // column block of this head inside each 192
    for (int mt = 0; mt < 4; ++mt) {
      v16h afr[6];
#pragma unroll
      for (int kt = 0; kt < 6; ++kt)
        afr[kt] = frag_rows(sX, C_, mt * 16 + l15, kt * 32, kb);
#pragma unroll
      for (int p = 0; p < 3; ++p) {  // 0=q, 1=k, 2=v
        const int n0g = p * C_ + wrow;
        const bool v1 = (16 + l15) < HD_;  // cols 24..31 are padding
        const float b0 = qkvb[n0g + l15];
        const float b1 = v1 ? qkvb[n0g + 16 + l15] : 0.0f;
        v8f acc0, acc1;
#pragma unroll
        for (int e = 0; e < 8; ++e) { acc0[e] = b0; acc1[e] = b1; }
#pragma unroll
        for (int kt = 0; kt < 6; ++kt) {
          v16h bf0 = frag_rows(Wqkv, C_, n0g + l15, kt * 32, kb);
          acc0 = wmma_f16(afr[kt], bf0, acc0);
          v16h bf1 = frag_rows_guard(Wqkv, C_, n0g + 16 + l15, kt * 32, kb, v1);
          acc1 = wmma_f16(afr[kt], bf1, acc1);
        }
        if (p < 2) {  // row L2-normalize over the 24 head dims
#pragma unroll
          for (int r = 0; r < 8; ++r) {
            float ss = acc0[r] * acc0[r] + acc1[r] * acc1[r];
#pragma unroll
            for (int off = 1; off < 16; off <<= 1)
              ss += __shfl_xor(ss, off, 32);
            float inv = 1.0f / fmaxf(sqrtf(ss), 1e-12f);
            if (p == 0) inv *= scale_h;  // fold logit scale into qn
            acc0[r] *= inv;
            acc1[r] *= inv;
          }
        }
        _Float16* dst = (p == 0) ? sQ[h] : (p == 1) ? sK[h] : sV[h];
#pragma unroll
        for (int r = 0; r < 8; ++r) {
          int m = mt * 16 + r + hi8;
          dst[m * 32 + l15]      = (_Float16)acc0[r];
          dst[m * 32 + 16 + l15] = (_Float16)acc1[r];
        }
      }
    }
  }
  __syncthreads();

  // ---- Phase C: attn = qn @ kn^T (+bias as WMMA C), softmax in-register ---
  {
    for (int mt = 0; mt < 4; ++mt) {
      v16h aq = frag_rows(sQ[h], 32, mt * 16 + l15, 0, kb);
      v8f at[4];
#pragma unroll
      for (int nt = 0; nt < 4; ++nt) {
        v16h bk = frag_rows(sK[h], 32, nt * 16 + l15, 0, kb);
        v8f c;
#pragma unroll
        for (int e = 0; e < 8; ++e) {
          int m = mt * 16 + e + hi8;
          c[e] = biasTab[((h * NT_) + m) * NT_ + nt * 16 + l15];
        }
        at[nt] = wmma_f16(aq, bk, c);
      }
#pragma unroll
      for (int r = 0; r < 8; ++r) {
        float mx = at[0][r];
#pragma unroll
        for (int nt = 1; nt < 4; ++nt) mx = fmaxf(mx, at[nt][r]);
#pragma unroll
        for (int off = 1; off < 16; off <<= 1)
          mx = fmaxf(mx, __shfl_xor(mx, off, 32));
        float ev[4];
        float s = 0.0f;
#pragma unroll
        for (int nt = 0; nt < 4; ++nt) {
          ev[nt] = __expf(at[nt][r] - mx);
          s += ev[nt];
        }
#pragma unroll
        for (int off = 1; off < 16; off <<= 1)
          s += __shfl_xor(s, off, 32);
        float inv = 1.0f / s;
        int m = mt * 16 + r + hi8;
#pragma unroll
        for (int nt = 0; nt < 4; ++nt)
          sP[h][m * NT_ + nt * 16 + l15] = (_Float16)(ev[nt] * inv);
      }
    }
  }
  __syncthreads();

  // ---- Phase D: head_out = P @ V, written into sX (input no longer needed)
  {
#pragma unroll
    for (int mt = 0; mt < 4; ++mt) {
#pragma unroll
      for (int nt = 0; nt < 2; ++nt) {
        v8f acc;
#pragma unroll
        for (int e = 0; e < 8; ++e) acc[e] = 0.0f;
#pragma unroll
        for (int kt = 0; kt < 2; ++kt) {
          v16h ap = frag_rows(sP[h], NT_, mt * 16 + l15, kt * 32, kb);
          v16h bv = frag_cols(sV[h], 32, kt * 32, nt * 16 + l15, kb);
          acc = wmma_f16(ap, bv, acc);
        }
        int n = nt * 16 + l15;
        if (n < HD_) {
#pragma unroll
          for (int r = 0; r < 8; ++r) {
            int m = mt * 16 + r + hi8;
            sX[m * C_ + h * HD_ + n] = (_Float16)acc[r];
          }
        }
      }
    }
  }
  __syncthreads();

  // ---- Phase E: proj GEMM + bias + residual (from LDS f32), scatter out ---
  {
    for (int tt = wave; tt < 48; tt += 8) {  // 4 M-tiles x 12 N-tiles
      int mt = tt / 12;
      int nt = tt % 12;
      int n0 = nt * 16;
      float pb = proj_b[n0 + l15];
      v8f acc;
#pragma unroll
      for (int e = 0; e < 8; ++e) acc[e] = pb;
#pragma unroll
      for (int kt = 0; kt < 6; ++kt) {
        v16h a  = frag_rows(sX, C_, mt * 16 + l15, kt * 32, kb);
        v16h bw = frag_rows(Wproj, C_, n0 + l15, kt * 32, kb);
        acc = wmma_f16(a, bw, acc);
      }
      int cch = n0 + l15;
#pragma unroll
      for (int e = 0; e < 8; ++e) {
        int m = mt * 16 + e + hi8;
        int i = m >> 3, j = m & 7;
        size_t gi =
            (((size_t)bb * C_ + cch) * HRES + row0 + i) * (size_t)WRES + col0 + j;
        out[gi] = sXf32[cch * NT_ + m] + acc[e];  // residual from LDS copy
      }
    }
  }
}

// ---------------------------------------------------------------------------
extern "C" void kernel_launch(void* const* d_in, const int* in_sizes, int n_in,
                              void* d_out, int out_size, void* d_ws,
                              size_t ws_size, hipStream_t stream) {
  (void)in_sizes; (void)n_in; (void)out_size; (void)ws_size;

  const float* x           = (const float*)d_in[0];
  // d_in[1] img_alpha: unused (no-op in reference)
  const float* qkv_w       = (const float*)d_in[2];
  const float* q_bias      = (const float*)d_in[3];
  const float* v_bias      = (const float*)d_in[4];
  const float* logit_scale = (const float*)d_in[5];
  const float* cpb_w1      = (const float*)d_in[6];
  const float* cpb_b1      = (const float*)d_in[7];
  const float* cpb_w2      = (const float*)d_in[8];
  const float* proj_w      = (const float*)d_in[9];
  const float* proj_b      = (const float*)d_in[10];

  char* ws = (char*)d_ws;
  float*    biasTab = (float*)(ws + OFF_BIAS);
  _Float16* WqkvH   = (_Float16*)(ws + OFF_WQKV);
  _Float16* WprojH  = (_Float16*)(ws + OFF_WPROJ);
  float*    qkvb    = (float*)(ws + OFF_QKVB);

  cpb_bias_kernel<<<1, 256, 0, stream>>>(cpb_w1, cpb_b1, cpb_w2, biasTab);

  const int total = 576 * 192 + 192 * 192 + 576;
  prep_weights_kernel<<<(total + 255) / 256, 256, 0, stream>>>(
      qkv_w, proj_w, q_bias, v_bias, WqkvH, WprojH, qkvb);

  winattn_main<<<4096, 256, 0, stream>>>(x, WqkvH, WprojH, qkvb, biasTab,
                                         logit_scale, proj_b, (float*)d_out);
}
